// MambaVisionMixer_36867999268916
// MI455X (gfx1250) — matrix operand order repaired and checked by reading, not compiled, not run on hardware
//
#include <hip/hip_runtime.h>
#include <hip/hip_bf16.h>

typedef __attribute__((ext_vector_type(16))) __bf16 v16bf;
typedef __attribute__((ext_vector_type(8)))  __bf16 v8bf;
typedef __attribute__((ext_vector_type(8)))  float  v8f;

// ---------------- bf16 split helpers (used only in the split kernel) -------
__device__ __forceinline__ unsigned short bf16_bits(float f) {
    unsigned u = __builtin_bit_cast(unsigned, f);
    return (unsigned short)((u + 0x7FFFu + ((u >> 16) & 1u)) >> 16);  // RNE
}
__device__ __forceinline__ float bf16_bits_to_f32(unsigned short h) {
    return __builtin_bit_cast(float, (unsigned)h << 16);
}
__device__ __forceinline__ __bf16 bits_to_bf16(unsigned short h) {
    return __builtin_bit_cast(__bf16, h);
}

// ---------------------------------------------------------------------------
// Elementwise fp32 -> (hi, lo) bf16 decomposition: f ~= hi + lo, so
// A*B ~= Ah*Bh + Ah*Bl + Al*Bh with fp32 accumulation (~fp32 accuracy).
// ---------------------------------------------------------------------------
__global__ __launch_bounds__(256)
void split_bf16_kernel(const float* __restrict__ src, __bf16* __restrict__ hi,
                       __bf16* __restrict__ lo, int n)
{
    const int i = blockIdx.x * 256 + threadIdx.x;
    if (i >= n) return;
    const float f = src[i];
    const unsigned short hb = bf16_bits(f);
    hi[i] = bits_to_bf16(hb);
    lo[i] = bits_to_bf16(bf16_bits(f - bf16_bits_to_f32(hb)));
}

// ---------------------------------------------------------------------------
// WMMA GEMM on pre-split bf16 operands: C[M,N] = A[M,K] * W[N,K]^T.
// Block: 256 threads = 8 waves; wave computes rows [m0,m0+16) x cols
// [n0,n0+64) as 4 accumulator tiles; grid = (N/64, M/128).
// Inner loop is pure b128 loads + 12x v_wmma per 32-wide K step.
// epilogue: 0 = plain store, 1 = v += 2*bias[col]; v = softplus(v)
// ---------------------------------------------------------------------------
__global__ __launch_bounds__(256)
void wmma_gemm_bf16_kernel(const __bf16* __restrict__ Ah, const __bf16* __restrict__ Al,
                           int lda,
                           const __bf16* __restrict__ Wh, const __bf16* __restrict__ Wl,
                           int ldw,
                           float* __restrict__ C, int ldc, int K,
                           const float* __restrict__ bias, int epilogue)
{
    const int lane = threadIdx.x & 31;
    const int wave = threadIdx.x >> 5;
    const int m0 = blockIdx.y * 128 + wave * 16;
    const int n0 = blockIdx.x * 64;

    v8f acc[4] = {};

    // A fragment: lane l -> row l&15;
    // element e at k0+koff+e, koff = 16*(l>=16).
    const int mrow   = m0 + (lane & 15);
    const int a_koff = (lane >> 4) * 16;
    const int b_koff = (lane >> 4) * 16;     // B: element e -> k0 + b_koff + e
    const __bf16* aph = Ah + (size_t)mrow * lda + a_koff;
    const __bf16* apl = Al + (size_t)mrow * lda + a_koff;

    for (int k0 = 0; k0 < K; k0 += 32) {
        const v8bf h0 = *(const v8bf*)(aph + k0);
        const v8bf h1 = *(const v8bf*)(aph + k0 + 8);
        const v8bf l0 = *(const v8bf*)(apl + k0);
        const v8bf l1 = *(const v8bf*)(apl + k0 + 8);
        const v16bf a_hi = __builtin_shufflevector(h0, h1,
            0, 1, 2, 3, 4, 5, 6, 7, 8, 9, 10, 11, 12, 13, 14, 15);
        const v16bf a_lo = __builtin_shufflevector(l0, l1,
            0, 1, 2, 3, 4, 5, 6, 7, 8, 9, 10, 11, 12, 13, 14, 15);

#pragma unroll
        for (int t = 0; t < 4; ++t) {
            const size_t wrow = (size_t)(n0 + t * 16 + (lane & 15)) * ldw + k0 + b_koff;
            const v16bf b_hi = *(const v16bf*)(Wh + wrow);
            const v16bf b_lo = *(const v16bf*)(Wl + wrow);
            acc[t] = __builtin_amdgcn_wmma_f32_16x16x32_bf16(
                         false, a_hi, false, b_hi, (short)0, acc[t], false, false);
            acc[t] = __builtin_amdgcn_wmma_f32_16x16x32_bf16(
                         false, a_hi, false, b_lo, (short)0, acc[t], false, false);
            acc[t] = __builtin_amdgcn_wmma_f32_16x16x32_bf16(
                         false, a_lo, false, b_hi, (short)0, acc[t], false, false);
        }
    }

    // C/D layout: VGPR r -> row m0 + r + 8*(lane>=16), col = n0 + t*16 + (lane&15)
    const int rbase = m0 + ((lane >> 4) * 8);
    const int clane = lane & 15;
#pragma unroll
    for (int t = 0; t < 4; ++t) {
        const int col = n0 + t * 16 + clane;
        const float bval = (epilogue == 1) ? 2.0f * bias[col] : 0.0f;
#pragma unroll
        for (int r = 0; r < 8; ++r) {
            float v = acc[t][r];
            if (epilogue == 1) {
                v += bval;
                v = (v > 20.0f) ? v : log1pf(__expf(v));   // softplus
            }
            C[(size_t)(rbase + r) * ldc + col] = v;
        }
    }
}

// ---------------------------------------------------------------------------
// Depthwise causal conv1d (K=4) + bias + SiLU.
// ---------------------------------------------------------------------------
__global__ __launch_bounds__(256)
void conv_silu_kernel(const float* __restrict__ xz, int col0,
                      const float* __restrict__ w, const float* __restrict__ b,
                      float* __restrict__ out, int out_ld, int out_col0)
{
    const int idx = blockIdx.x * blockDim.x + threadIdx.x;   // M*512 threads
    const int row = idx >> 9;          // b*4096 + l
    const int d   = idx & 511;
    const int l   = row & 4095;

    const float* src = xz + (size_t)row * 1024 + col0 + d;
    float acc = b[d];
    const float w0 = w[d * 4 + 0], w1 = w[d * 4 + 1],
                w2 = w[d * 4 + 2], w3 = w[d * 4 + 3];
    if (l >= 3) acc += w0 * src[-3 * 1024];
    if (l >= 2) acc += w1 * src[-2 * 1024];
    if (l >= 1) acc += w2 * src[-1 * 1024];
    acc += w3 * src[0];
    const float s = acc * (1.0f / (1.0f + __expf(-acc)));    // silu
    out[(size_t)row * out_ld + out_col0 + d] = s;
}

// ---------------------------------------------------------------------------
// Selective scan: 16 lanes per channel (one per state), shfl_xor reduction.
// ---------------------------------------------------------------------------
__global__ __launch_bounds__(256)
void scan_kernel(const float* __restrict__ delta, const float* __restrict__ xc,
                 const float* __restrict__ xdbl,
                 const float* __restrict__ A_log, const float* __restrict__ Dp,
                 float* __restrict__ ycat)
{
    const int grp = threadIdx.x >> 4;
    const int n   = threadIdx.x & 15;
    const int c   = blockIdx.x * 16 + grp;      // 0..1023
    const int b   = c >> 9;
    const int d   = c & 511;

    const float Acoef = -__expf(A_log[d * 16 + n]);
    const float Dd    = Dp[d];
    float h = 0.0f;
    const size_t rbase = (size_t)b * 4096;

    for (int l = 0; l < 4096; ++l) {
        const size_t row = rbase + l;
        const float dlt = delta[row * 512 + d];
        const float u   = xc[row * 512 + d];
        const float Bn  = xdbl[row * 64 + 32 + n];
        const float Cn  = xdbl[row * 64 + 48 + n];

        h = __expf(dlt * Acoef) * h + (dlt * u) * Bn;

        float p = h * Cn;
        p += __shfl_xor(p, 1, 32);
        p += __shfl_xor(p, 2, 32);
        p += __shfl_xor(p, 4, 32);
        p += __shfl_xor(p, 8, 32);
        if (n == 0) ycat[row * 1024 + d] = p + Dd * u;
    }
}

// ---------------------------------------------------------------------------
static char* ws_take(char*& p, size_t bytes) {
    char* r = p;
    p += (bytes + 255) & ~(size_t)255;
    return r;
}

extern "C" void kernel_launch(void* const* d_in, const int* in_sizes, int n_in,
                              void* d_out, int out_size, void* d_ws, size_t ws_size,
                              hipStream_t stream)
{
    const float* hidden   = (const float*)d_in[0];   // [2,4096,512]
    const float* W_in     = (const float*)d_in[1];   // [1024,512]
    const float* W_out    = (const float*)d_in[2];   // [512,1024]
    const float* conv_x_w = (const float*)d_in[3];
    const float* conv_x_b = (const float*)d_in[4];
    const float* conv_z_w = (const float*)d_in[5];
    const float* conv_z_b = (const float*)d_in[6];
    const float* W_xproj  = (const float*)d_in[7];   // [64,512]
    const float* W_dt     = (const float*)d_in[8];   // [512,32]
    const float* b_dt     = (const float*)d_in[9];
    const float* A_log    = (const float*)d_in[10];  // [512,16]
    const float* Dp       = (const float*)d_in[11];
    float* out = (float*)d_out;                      // [2,4096,512]

    const int M = 8192;                              // B*L

    // ---- workspace layout ----
    char* p = (char*)d_ws;
    float* xz_f    = (float*)ws_take(p, (size_t)M * 1024 * 4);
    float* xc_f    = (float*)ws_take(p, (size_t)M * 512 * 4);
    float* xdbl_f  = (float*)ws_take(p, (size_t)M * 64 * 4);
    float* delta_f = (float*)ws_take(p, (size_t)M * 512 * 4);
    float* ycat_f  = (float*)ws_take(p, (size_t)M * 1024 * 4);

    // regionA: hidden split (alive only for GEMM1) overlaid by ycat split
    char* regionA = ws_take(p, (size_t)M * 1024 * 2 * 2);
    __bf16* hid_h  = (__bf16*)regionA;
    __bf16* hid_l  = hid_h + (size_t)M * 512;
    __bf16* ycat_h = (__bf16*)regionA;
    __bf16* ycat_l = ycat_h + (size_t)M * 1024;

    __bf16* xc_h   = (__bf16*)ws_take(p, (size_t)M * 512 * 2 * 2);
    __bf16* xc_l   = xc_h + (size_t)M * 512;
    __bf16* xdbl_h = (__bf16*)ws_take(p, (size_t)M * 64 * 2 * 2);
    __bf16* xdbl_l = xdbl_h + (size_t)M * 64;

    __bf16* win_h  = (__bf16*)ws_take(p, (size_t)1024 * 512 * 2 * 2);
    __bf16* win_l  = win_h + 1024 * 512;
    __bf16* wxp_h  = (__bf16*)ws_take(p, (size_t)64 * 512 * 2 * 2);
    __bf16* wxp_l  = wxp_h + 64 * 512;
    __bf16* wdt_h  = (__bf16*)ws_take(p, (size_t)512 * 32 * 2 * 2);
    __bf16* wdt_l  = wdt_h + 512 * 32;
    __bf16* wout_h = (__bf16*)ws_take(p, (size_t)512 * 1024 * 2 * 2);
    __bf16* wout_l = wout_h + 512 * 1024;

    dim3 blk(256);
    auto grids = [](int n) { return dim3((n + 255) / 256); };

    // ---- weight splits (tiny) ----
    split_bf16_kernel<<<grids(1024 * 512), blk, 0, stream>>>(W_in,    win_h,  win_l,  1024 * 512);
    split_bf16_kernel<<<grids(64 * 512),   blk, 0, stream>>>(W_xproj, wxp_h,  wxp_l,  64 * 512);
    split_bf16_kernel<<<grids(512 * 32),   blk, 0, stream>>>(W_dt,    wdt_h,  wdt_l,  512 * 32);
    split_bf16_kernel<<<grids(512 * 1024), blk, 0, stream>>>(W_out,   wout_h, wout_l, 512 * 1024);

    // 1) xz = hidden @ W_in^T      (8192 x 1024 x 512)
    split_bf16_kernel<<<grids(M * 512), blk, 0, stream>>>(hidden, hid_h, hid_l, M * 512);
    wmma_gemm_bf16_kernel<<<dim3(1024 / 64, M / 128), blk, 0, stream>>>(
        hid_h, hid_l, 512, win_h, win_l, 512, xz_f, 1024, 512, nullptr, 0);

    // 2) xc = silu(conv(x)); zc = silu(conv(z)) -> ycat[:,512:]
    conv_silu_kernel<<<(M * 512) / 256, blk, 0, stream>>>(
        xz_f, 0,   conv_x_w, conv_x_b, xc_f,   512,  0);
    conv_silu_kernel<<<(M * 512) / 256, blk, 0, stream>>>(
        xz_f, 512, conv_z_w, conv_z_b, ycat_f, 1024, 512);

    // 3) x_dbl = xc @ W_xproj^T    (8192 x 64 x 512)
    split_bf16_kernel<<<grids(M * 512), blk, 0, stream>>>(xc_f, xc_h, xc_l, M * 512);
    wmma_gemm_bf16_kernel<<<dim3(64 / 64, M / 128), blk, 0, stream>>>(
        xc_h, xc_l, 512, wxp_h, wxp_l, 512, xdbl_f, 64, 512, nullptr, 0);

    // 4) delta = softplus(dt @ W_dt^T + 2*b_dt)   (8192 x 512 x 32)
    split_bf16_kernel<<<grids(M * 64), blk, 0, stream>>>(xdbl_f, xdbl_h, xdbl_l, M * 64);
    wmma_gemm_bf16_kernel<<<dim3(512 / 64, M / 128), blk, 0, stream>>>(
        xdbl_h, xdbl_l, 64, wdt_h, wdt_l, 32, delta_f, 512, 32, b_dt, 1);

    // 5) selective scan -> ycat[:, :512]
    scan_kernel<<<64, blk, 0, stream>>>(delta_f, xc_f, xdbl_f, A_log, Dp, ycat_f);

    // 6) out = ycat @ W_out^T      (8192 x 512 x 1024)
    split_bf16_kernel<<<grids(M * 1024), blk, 0, stream>>>(ycat_f, ycat_h, ycat_l, M * 1024);
    wmma_gemm_bf16_kernel<<<dim3(512 / 64, M / 128), blk, 0, stream>>>(
        ycat_h, ycat_l, 1024, wout_h, wout_l, 1024, out, 512, 1024, nullptr, 0);
}
